// HypergraphConv_7541962571982
// MI455X (gfx1250) — matrix operand order repaired and checked by reading, not compiled, not run on hardware
//
#include <hip/hip_runtime.h>
#include <stdint.h>

// Problem constants (from reference)
#define BB 4
#define NN 2048
#define EE 4096
#define CC 64
#define EPSF 1e-6f

typedef __attribute__((ext_vector_type(16))) __bf16        v16bf;
typedef __attribute__((ext_vector_type(8)))  float         v8f;
typedef __attribute__((ext_vector_type(8)))  unsigned int  v8u;

__device__ __forceinline__ uint16_t f32_to_bf16_rne(float f) {
    unsigned u = __float_as_uint(f);
    u += 0x7FFFu + ((u >> 16) & 1u);
    return (uint16_t)(u >> 16);
}
__device__ __forceinline__ float bf16u_to_f32(uint16_t h) {
    return __uint_as_float(((unsigned)h) << 16);
}

// ---------------------------------------------------------------------------
// K1: node degrees -> u[b,n] = 1/sqrt(sum_e H[b,n,e])   (one wave per row)
// ---------------------------------------------------------------------------
__global__ void k_node_deg(const float* __restrict__ H, float* __restrict__ u) {
    int wave = (blockIdx.x * blockDim.x + threadIdx.x) >> 5;  // b*N + n
    int lane = threadIdx.x & 31;
    if (wave >= BB * NN) return;
    const float* row = H + (size_t)wave * EE;
    float s = 0.f;
    for (int e = lane; e < EE; e += 32) s += row[e];
    #pragma unroll
    for (int off = 16; off; off >>= 1) s += __shfl_down(s, off, 32);
    if (lane == 0) u[wave] = (s > 0.f) ? (1.0f / sqrtf(s)) : 0.f;
}

// ---------------------------------------------------------------------------
// K2: edge degrees -> v[b,e] = 1/sum_n H[b,n,e]
// block: 256 threads = 64 e-columns x 4 n-slices (fixed-order -> deterministic)
// ---------------------------------------------------------------------------
__global__ void k_edge_deg(const float* __restrict__ H, float* __restrict__ v) {
    __shared__ float s[256];
    int b  = blockIdx.x / (EE / 64);
    int eg = blockIdx.x % (EE / 64);
    int el = threadIdx.x & 63;
    int sl = threadIdx.x >> 6;
    int e  = eg * 64 + el;
    const float* base = H + ((size_t)b * NN) * EE + e;
    float acc = 0.f;
    int n0 = sl * (NN / 4);
    for (int n = n0; n < n0 + NN / 4; ++n) acc += base[(size_t)n * EE];
    s[threadIdx.x] = acc;
    __syncthreads();
    if (threadIdx.x < 64) {
        float d = s[el] + s[el + 64] + s[el + 128] + s[el + 192];
        v[b * EE + e] = (d > 0.f) ? (1.0f / d) : 0.f;
    }
}

// ---------------------------------------------------------------------------
// K3: t[b] = sum_n u[b,n] ; r[b] = sum_e v[b,e]   (deterministic tree)
// ---------------------------------------------------------------------------
__global__ void k_sums(const float* __restrict__ u, const float* __restrict__ v,
                       float* __restrict__ t, float* __restrict__ r) {
    __shared__ float s[256];
    int b = blockIdx.x & 3, mode = blockIdx.x >> 2;
    const float* src = mode ? (v + (size_t)b * EE) : (u + (size_t)b * NN);
    int len = mode ? EE : NN;
    float acc = 0.f;
    for (int i = threadIdx.x; i < len; i += 256) acc += src[i];
    s[threadIdx.x] = acc;
    __syncthreads();
    for (int o = 128; o; o >>= 1) {
        if (threadIdx.x < o) s[threadIdx.x] += s[threadIdx.x + o];
        __syncthreads();
    }
    if (threadIdx.x == 0) (mode ? r : t)[b] = s[0];
}

// ---------------------------------------------------------------------------
// K4: Y = x @ W   [B*N,64] x [64,64]
// ---------------------------------------------------------------------------
__global__ void k_xw(const float* __restrict__ x, const float* __restrict__ w,
                     float* __restrict__ Y) {
    int idx = blockIdx.x * 256 + threadIdx.x;     // (b*N+n)*64 + c
    int row = idx >> 6, c = idx & 63;
    const float* xr = x + (size_t)row * CC;
    float acc = 0.f;
    #pragma unroll
    for (int k = 0; k < CC; ++k) acc += xr[k] * w[k * CC + c];
    Y[idx] = acc;
}

// ---------------------------------------------------------------------------
// K5/K11: out[b,c] = sum_i wgt[b,i] * M[b,i,c]   (rank-1 correction vectors)
// ---------------------------------------------------------------------------
__global__ void k_wcolsum(const float* __restrict__ wgt, const float* __restrict__ M,
                          float* __restrict__ out, int len) {
    __shared__ float s[256];
    int b = blockIdx.x >> 6, c = blockIdx.x & 63;
    const float* wb = wgt + (size_t)b * len;
    const float* Mb = M + (size_t)b * len * CC + c;
    float acc = 0.f;
    for (int i = threadIdx.x; i < len; i += 256) acc += wb[i] * Mb[(size_t)i * CC];
    s[threadIdx.x] = acc;
    __syncthreads();
    for (int o = 128; o; o >>= 1) {
        if (threadIdx.x < o) s[threadIdx.x] += s[threadIdx.x + o];
        __syncthreads();
    }
    if (threadIdx.x == 0) out[blockIdx.x] = s[0];
}

// ---------------------------------------------------------------------------
// K6: Z1 = Dv^{-1/2} Y  = diag(u)Y - cv * u * p ; store TRANSPOSED bf16 hi/lo
//     Z1t[b][c][n]
// ---------------------------------------------------------------------------
__global__ void k_z1(const float* __restrict__ u, const float* __restrict__ Y,
                     const float* __restrict__ p, const float* __restrict__ t,
                     uint16_t* __restrict__ Zh, uint16_t* __restrict__ Zl) {
    int idx = blockIdx.x * 256 + threadIdx.x;     // (b*64+c)*N + n
    int n  = idx & (NN - 1);
    int bc = idx >> 11;                            // b*64 + c
    int b  = bc >> 6;
    float cv = EPSF / (1.f + EPSF * t[b]);
    float un = u[b * NN + n];
    float y  = Y[((size_t)(b * NN + n)) * CC + (bc & 63)];
    float z  = un * y - cv * un * p[bc];
    uint16_t hi = f32_to_bf16_rne(z);
    Zh[idx] = hi;
    Zl[idx] = f32_to_bf16_rne(z - bf16u_to_f32(hi));
}

// ---------------------------------------------------------------------------
// K7: GEMM1  Z2t[b][c][e] = sum_n Z1[n,c] * H[n,e]
//     WMMA: A[m=c][k=n] = Z1t rows (contiguous), B[k=n][j=e] = H rows.
//     H is {0,1} -> bf16 by exact bit-truncation. hi/lo split on A.
//     One wave per (b, cTile, eTile): 4*4*256 = 4096 waves.
// ---------------------------------------------------------------------------
__global__ void k_gemm1(const float* __restrict__ H, const uint16_t* __restrict__ Zh,
                        const uint16_t* __restrict__ Zl, float* __restrict__ Z2) {
    int wg   = (blockIdx.x * blockDim.x + threadIdx.x) >> 5;
    int lane = threadIdx.x & 31;
    int b  = wg >> 10, rem = wg & 1023;
    int c0 = (rem >> 8) * 16, e0 = (rem & 255) * 16;
    int m = lane & 15, g = lane >> 4;

    const uint16_t* Arh = Zh + ((size_t)(b * CC + c0 + m)) * NN;
    const uint16_t* Arl = Zl + ((size_t)(b * CC + c0 + m)) * NN;
    const float*    Hb  = H + ((size_t)b * NN) * EE + e0 + m;   // col j = lane&15

    v8f acc = {0.f, 0.f, 0.f, 0.f, 0.f, 0.f, 0.f, 0.f};

    for (int n0 = 0; n0 < NN; n0 += 32) {
        // A (hi): lane holds K = 8g..8g+7 and 16+8g..16+8g+7 (two 16B runs)
        uint4 a0 = *(const uint4*)(Arh + n0 + 8 * g);
        uint4 a1 = *(const uint4*)(Arh + n0 + 16 + 8 * g);
        v8u pk;
        pk[0] = a0.x; pk[1] = a0.y; pk[2] = a0.z; pk[3] = a0.w;
        pk[4] = a1.x; pk[5] = a1.y; pk[6] = a1.z; pk[7] = a1.w;
        v16bf Ahi = __builtin_bit_cast(v16bf, pk);
        a0 = *(const uint4*)(Arl + n0 + 8 * g);
        a1 = *(const uint4*)(Arl + n0 + 16 + 8 * g);
        pk[0] = a0.x; pk[1] = a0.y; pk[2] = a0.z; pk[3] = a0.w;
        pk[4] = a1.x; pk[5] = a1.y; pk[6] = a1.z; pk[7] = a1.w;
        v16bf Alo = __builtin_bit_cast(v16bf, pk);

        // B: VGPR i holds rows K = 8g+2i, 8g+2i+1 (i<4), 16+8g+2(i-4).. (i>=4) at column j
        const float* Hp = Hb + (size_t)(n0 + 8 * g) * EE;
        v8u bp;
        #pragma unroll
        for (int i = 0; i < 8; ++i) {
            unsigned f0 = __float_as_uint(Hp[(size_t)(2 * i + (i >= 4 ? 8 : 0)) * EE]);
            unsigned f1 = __float_as_uint(Hp[(size_t)(2 * i + 1 + (i >= 4 ? 8 : 0)) * EE]);
            bp[i] = (f0 >> 16) | (f1 & 0xFFFF0000u);   // exact for {0,1}
        }
        v16bf Bf = __builtin_bit_cast(v16bf, bp);

        acc = __builtin_amdgcn_wmma_f32_16x16x32_bf16(false, Ahi, false, Bf,
                                                      (short)0, acc, false, false);
        acc = __builtin_amdgcn_wmma_f32_16x16x32_bf16(false, Alo, false, Bf,
                                                      (short)0, acc, false, false);
    }

    // D layout: VGPR r -> (M = r + 8*(lane>>4), N = lane&15)
    int nCol = lane & 15, mOff = (lane >> 4) * 8;
    float* Zo = Z2 + ((size_t)(b * CC + c0 + mOff)) * EE + e0 + nCol;
    #pragma unroll
    for (int rr = 0; rr < 8; ++rr) Zo[(size_t)rr * EE] = acc[rr];
}

// ---------------------------------------------------------------------------
// K8: q[b,c] = sum_e v[b,e] * Z2t[b][c][e]
// ---------------------------------------------------------------------------
__global__ void k_q(const float* __restrict__ v, const float* __restrict__ Z2,
                    float* __restrict__ q) {
    __shared__ float s[256];
    int b = blockIdx.x >> 6, c = blockIdx.x & 63;
    const float* vb = v + (size_t)b * EE;
    const float* z  = Z2 + ((size_t)(b * CC + c)) * EE;
    float acc = 0.f;
    for (int e = threadIdx.x; e < EE; e += 256) acc += vb[e] * z[e];
    s[threadIdx.x] = acc;
    __syncthreads();
    for (int o = 128; o; o >>= 1) {
        if (threadIdx.x < o) s[threadIdx.x] += s[threadIdx.x + o];
        __syncthreads();
    }
    if (threadIdx.x == 0) q[blockIdx.x] = s[0];
}

// ---------------------------------------------------------------------------
// K9: Z3[b][e][c] = De^{-1} Z2 = v*Z2 - ce*v*q ; bf16 hi/lo, row-major [e][c]
// ---------------------------------------------------------------------------
__global__ void k_z3(const float* __restrict__ v, const float* __restrict__ Z2,
                     const float* __restrict__ q, const float* __restrict__ r,
                     uint16_t* __restrict__ Zh, uint16_t* __restrict__ Zl) {
    int idx = blockIdx.x * 256 + threadIdx.x;     // (b*E+e)*64 + c
    int c   = idx & 63;
    int row = idx >> 6;                            // b*E + e
    int b   = row >> 12;
    int e   = row & (EE - 1);
    float ce = EPSF / (1.f + EPSF * r[b]);
    float ve = v[row];
    float z2 = Z2[((size_t)(b * CC + c)) * EE + e];
    float z  = ve * z2 - ce * ve * q[b * CC + c];
    uint16_t hi = f32_to_bf16_rne(z);
    Zh[idx] = hi;
    Zl[idx] = f32_to_bf16_rne(z - bf16u_to_f32(hi));
}

// ---------------------------------------------------------------------------
// K10: GEMM2  Z4[b][n][c] = sum_e H[n,e] * Z3[e,c]
//      A[m=n][k=e] = H rows (contiguous, bit-truncated to bf16),
//      B[k=e][j=c] = Z3 rows, hi/lo split on B.
//      One wave per (b, nTile, cTile): 4*128*4 = 2048 waves.
// ---------------------------------------------------------------------------
__global__ void k_gemm2(const float* __restrict__ H, const uint16_t* __restrict__ Zh,
                        const uint16_t* __restrict__ Zl, float* __restrict__ Z4) {
    int wg   = (blockIdx.x * blockDim.x + threadIdx.x) >> 5;
    int lane = threadIdx.x & 31;
    int b  = wg >> 9, rem = wg & 511;
    int n0 = (rem >> 2) * 16, c0 = (rem & 3) * 16;
    int m = lane & 15, g = lane >> 4;

    const unsigned*  Hr  = (const unsigned*)(H + ((size_t)(b * NN + n0 + m)) * EE);
    const uint16_t*  Bhb = Zh + ((size_t)b * EE) * CC + c0 + m;   // col j = lane&15
    const uint16_t*  Blb = Zl + ((size_t)b * EE) * CC + c0 + m;

    v8f acc = {0.f, 0.f, 0.f, 0.f, 0.f, 0.f, 0.f, 0.f};

    for (int e0 = 0; e0 < EE; e0 += 32) {
        uint4 f0 = *(const uint4*)(Hr + e0 + 16 * g);
        uint4 f1 = *(const uint4*)(Hr + e0 + 16 * g + 4);
        uint4 f2 = *(const uint4*)(Hr + e0 + 16 * g + 8);
        uint4 f3 = *(const uint4*)(Hr + e0 + 16 * g + 12);
        v8u pk;
        pk[0] = (f0.x >> 16) | (f0.y & 0xFFFF0000u);
        pk[1] = (f0.z >> 16) | (f0.w & 0xFFFF0000u);
        pk[2] = (f1.x >> 16) | (f1.y & 0xFFFF0000u);
        pk[3] = (f1.z >> 16) | (f1.w & 0xFFFF0000u);
        pk[4] = (f2.x >> 16) | (f2.y & 0xFFFF0000u);
        pk[5] = (f2.z >> 16) | (f2.w & 0xFFFF0000u);
        pk[6] = (f3.x >> 16) | (f3.y & 0xFFFF0000u);
        pk[7] = (f3.z >> 16) | (f3.w & 0xFFFF0000u);
        v16bf Af = __builtin_bit_cast(v16bf, pk);

        const uint16_t* Bh = Bhb + (size_t)(e0 + 16 * g) * CC;
        const uint16_t* Bl = Blb + (size_t)(e0 + 16 * g) * CC;
        v8u bh, bl;
        #pragma unroll
        for (int i = 0; i < 8; ++i) {
            bh[i] = (unsigned)Bh[(size_t)(2 * i) * CC] |
                    ((unsigned)Bh[(size_t)(2 * i + 1) * CC] << 16);
            bl[i] = (unsigned)Bl[(size_t)(2 * i) * CC] |
                    ((unsigned)Bl[(size_t)(2 * i + 1) * CC] << 16);
        }
        acc = __builtin_amdgcn_wmma_f32_16x16x32_bf16(false, Af, false,
                  __builtin_bit_cast(v16bf, bh), (short)0, acc, false, false);
        acc = __builtin_amdgcn_wmma_f32_16x16x32_bf16(false, Af, false,
                  __builtin_bit_cast(v16bf, bl), (short)0, acc, false, false);
    }

    int cCol = lane & 15, mOff = (lane >> 4) * 8;
    float* Zo = Z4 + ((size_t)(b * NN + n0 + mOff)) * CC + c0 + cCol;
    #pragma unroll
    for (int rr = 0; rr < 8; ++rr) Zo[(size_t)rr * CC] = acc[rr];
}

// ---------------------------------------------------------------------------
// K12: out = diag(u) Z4 - cv * u * p2 + bias
// ---------------------------------------------------------------------------
__global__ void k_out(const float* __restrict__ u, const float* __restrict__ Z4,
                      const float* __restrict__ p2, const float* __restrict__ t,
                      const float* __restrict__ bias, float* __restrict__ out) {
    int idx = blockIdx.x * 256 + threadIdx.x;     // (b*N+n)*64 + c
    int c   = idx & 63;
    int row = idx >> 6;
    int b   = row >> 11;
    float cv = EPSF / (1.f + EPSF * t[b]);
    float un = u[row];
    out[idx] = un * Z4[idx] - cv * un * p2[b * CC + c] + bias[c];
}

// ---------------------------------------------------------------------------
// Workspace layout (bytes)
// ---------------------------------------------------------------------------
constexpr size_t OFF_U   = 0;                                     // B*N f32
constexpr size_t OFF_V   = OFF_U + (size_t)BB * NN * 4;           // B*E f32
constexpr size_t OFF_T   = OFF_V + (size_t)BB * EE * 4;           // B f32
constexpr size_t OFF_R   = OFF_T + 64;
constexpr size_t OFF_P   = OFF_R + 64;                            // B*64 f32
constexpr size_t OFF_Q   = OFF_P + (size_t)BB * CC * 4;
constexpr size_t OFF_P2  = OFF_Q + (size_t)BB * CC * 4;
constexpr size_t OFF_Y   = ((OFF_P2 + (size_t)BB * CC * 4 + 255) / 256) * 256;
constexpr size_t OFF_Z1H = OFF_Y + (size_t)BB * NN * CC * 4;      // bf16 [b][c][n]
constexpr size_t OFF_Z1L = OFF_Z1H + (size_t)BB * CC * NN * 2;
constexpr size_t OFF_Z2  = OFF_Z1L + (size_t)BB * CC * NN * 2;    // f32 [b][c][e]
constexpr size_t OFF_Z3H = OFF_Z2 + (size_t)BB * CC * EE * 4;     // bf16 [b][e][c]
constexpr size_t OFF_Z3L = OFF_Z3H + (size_t)BB * EE * CC * 2;
constexpr size_t OFF_Z4  = OFF_Z3L + (size_t)BB * EE * CC * 2;    // f32 [b][n][c]

extern "C" void kernel_launch(void* const* d_in, const int* in_sizes, int n_in,
                              void* d_out, int out_size, void* d_ws, size_t ws_size,
                              hipStream_t stream) {
    const float* x    = (const float*)d_in[0];
    const float* H    = (const float*)d_in[1];
    const float* w    = (const float*)d_in[2];
    const float* bias = (const float*)d_in[3];
    float* out = (float*)d_out;

    char* ws = (char*)d_ws;
    float*    u   = (float*)(ws + OFF_U);
    float*    v   = (float*)(ws + OFF_V);
    float*    t   = (float*)(ws + OFF_T);
    float*    r   = (float*)(ws + OFF_R);
    float*    p   = (float*)(ws + OFF_P);
    float*    q   = (float*)(ws + OFF_Q);
    float*    p2  = (float*)(ws + OFF_P2);
    float*    Y   = (float*)(ws + OFF_Y);
    uint16_t* Z1h = (uint16_t*)(ws + OFF_Z1H);
    uint16_t* Z1l = (uint16_t*)(ws + OFF_Z1L);
    float*    Z2  = (float*)(ws + OFF_Z2);
    uint16_t* Z3h = (uint16_t*)(ws + OFF_Z3H);
    uint16_t* Z3l = (uint16_t*)(ws + OFF_Z3L);
    float*    Z4  = (float*)(ws + OFF_Z4);

    k_node_deg<<<dim3(BB * NN / 8), 256, 0, stream>>>(H, u);
    k_edge_deg<<<dim3(BB * EE / 64), 256, 0, stream>>>(H, v);
    k_sums    <<<dim3(2 * BB), 256, 0, stream>>>(u, v, t, r);
    k_xw      <<<dim3(BB * NN * CC / 256), 256, 0, stream>>>(x, w, Y);
    k_wcolsum <<<dim3(BB * CC), 256, 0, stream>>>(u, Y, p, NN);
    k_z1      <<<dim3(BB * CC * NN / 256), 256, 0, stream>>>(u, Y, p, t, Z1h, Z1l);
    k_gemm1   <<<dim3(BB * 4 * (EE / 16) / 8), 256, 0, stream>>>(H, Z1h, Z1l, Z2);
    k_q       <<<dim3(BB * CC), 256, 0, stream>>>(v, Z2, q);
    k_z3      <<<dim3(BB * EE * CC / 256), 256, 0, stream>>>(v, Z2, q, r, Z3h, Z3l);
    k_gemm2   <<<dim3(BB * (NN / 16) * 4 / 8), 256, 0, stream>>>(H, Z3h, Z3l, Z4);
    k_wcolsum <<<dim3(BB * CC), 256, 0, stream>>>(u, Z4, p2, NN);
    k_out     <<<dim3(BB * NN * CC / 256), 256, 0, stream>>>(u, Z4, p2, t, bias, out);
}
